// Block_22686017258100
// MI455X (gfx1250) — matrix operand order repaired and checked by reading, not compiled, not run on hardware
//
#include <hip/hip_runtime.h>
#include <hip/hip_bf16.h>
#include <math.h>

#define B_  8
#define T_  1024
#define C_  1024
#define H_  16
#define D_  64
#define TI_ 256
#define FF_ 4096

typedef __bf16 bf16;
typedef __attribute__((ext_vector_type(16))) __bf16 bf16x16;
typedef __attribute__((ext_vector_type(8)))  __bf16 bf16x8;
typedef __attribute__((ext_vector_type(8)))  float  f32x8;

// Build a 16-element bf16 WMMA fragment from two contiguous 8-element (16B) chunks.
__device__ __forceinline__ bf16x16 make_frag(const bf16* p0, const bf16* p1) {
    bf16x8 lo = *reinterpret_cast<const bf16x8*>(p0);
    bf16x8 hi = *reinterpret_cast<const bf16x8*>(p1);
    bf16x16 r;
#pragma unroll
    for (int i = 0; i < 8; ++i) { r[i] = lo[i]; r[i + 8] = hi[i]; }
    return r;
}

__device__ __forceinline__ f32x8 wmma_bf16(bf16x16 a, bf16x16 b, f32x8 c) {
    return __builtin_amdgcn_wmma_f32_16x16x32_bf16(
        /*neg_a=*/false, a, /*neg_b=*/false, b,
        /*c_mod=*/(short)0, c, /*reuse_a=*/false, /*reuse_b=*/false);
}

// Async global->LDS 16B copy (no VGPR round trip; tracked by ASYNCcnt).
// VDST VGPR carries the LDS byte offset: low 32 bits of a generic __shared__
// pointer are the AS(3) offset (aperture base lives in addr[63:32]).
__device__ __forceinline__ void async_copy16(const void* gptr, void* lptr) {
    uint32_t l = (uint32_t)(size_t)lptr;
    uint64_t g = (uint64_t)(size_t)gptr;
    asm volatile("global_load_async_to_lds_b128 %0, %1, off" :: "v"(l), "v"(g) : "memory");
}
__device__ __forceinline__ void wait_async0() {
    asm volatile("s_wait_asynccnt 0x0" ::: "memory");
}

__device__ __forceinline__ float gelu_tanh(float x) {
    float x3 = x * x * x;
    return 0.5f * x * (1.0f + tanhf(0.7978845608028654f * (x + 0.044715f * x3)));
}

// ---------------------------------------------------------------------------
// Tiled bf16 GEMM: out[M,N] = epilogue(A[M,K] @ Bt[N,K]^T + bias)
// Async double-buffered LDS staging: DMA for tile k+1 overlaps WMMAs of tile k.
// flags: 1 = GELU, 2 = add f32 residual, 4 = write f32 (else bf16)
// Requires M%128==0, N%128==0, K%32==0.
// ---------------------------------------------------------------------------
__global__ __launch_bounds__(256)
void gemm_bf16_wmma(const bf16* __restrict__ A, const bf16* __restrict__ Bt,
                    const float* __restrict__ bias, const float* __restrict__ resid,
                    float* __restrict__ outF, bf16* __restrict__ outB,
                    int M, int N, int K, int flags)
{
    __shared__ bf16 As[2][128][40];   // 32 K + 8 pad (80B pitch, 16B aligned)
    __shared__ bf16 Bs[2][128][40];

    const int tid   = threadIdx.x;
    const int w     = tid >> 5, lane = tid & 31;
    const int laneM = lane & 15, laneH = lane >> 4;
    const int wr    = w >> 1,   wc   = w & 1;       // 4x2 wave grid
    const int m0    = blockIdx.y * 128;
    const int n0    = blockIdx.x * 128;

    f32x8 acc[2][4];
#pragma unroll
    for (int i = 0; i < 2; ++i)
#pragma unroll
        for (int j = 0; j < 4; ++j)
#pragma unroll
            for (int r = 0; r < 8; ++r) acc[i][j][r] = 0.0f;

    const int lr   = tid >> 1;   // 0..127 : tile row
    const int half = tid & 1;    // which pair of 16B chunks in the 64B row slab

    // Issue async DMA for one 128x32 A tile + one 128x32 B tile into buffer `buf`.
    auto stage = [&](int buf, int k0) {
#pragma unroll
        for (int j = 0; j < 2; ++j) {
            const int c = half * 2 + j;  // 16B chunk 0..3 within the row's 64B
            async_copy16(A  + (size_t)(m0 + lr) * K + k0 + c * 8, &As[buf][lr][c * 8]);
            async_copy16(Bt + (size_t)(n0 + lr) * K + k0 + c * 8, &Bs[buf][lr][c * 8]);
        }
    };

    const int nk = K >> 5;
    stage(0, 0);

    for (int i = 0; i < nk; ++i) {
        const int cur = i & 1;
        wait_async0();        // own DMA for `cur` complete
        __syncthreads();      // everyone's DMA complete + prior reads of cur^1 done
        if (i + 1 < nk) stage(cur ^ 1, (i + 1) * 32);

        bf16x16 af[2], bfg[4];
#pragma unroll
        for (int mt = 0; mt < 2; ++mt) {
            const bf16* base = &As[cur][wr * 32 + mt * 16 + laneM][0];
            af[mt] = make_frag(base + laneH * 8, base + 16 + laneH * 8);
        }
#pragma unroll
        for (int nt = 0; nt < 4; ++nt) {
            const bf16* base = &Bs[cur][wc * 64 + nt * 16 + laneM][0];
            bfg[nt] = make_frag(base + laneH * 8, base + 16 + laneH * 8);
        }
#pragma unroll
        for (int mt = 0; mt < 2; ++mt)
#pragma unroll
            for (int nt = 0; nt < 4; ++nt)
                acc[mt][nt] = wmma_bf16(af[mt], bfg[nt], acc[mt][nt]);
    }

#pragma unroll
    for (int mt = 0; mt < 2; ++mt) {
#pragma unroll
        for (int nt = 0; nt < 4; ++nt) {
            const int col = n0 + wc * 64 + nt * 16 + laneM;
            const float bv = bias ? bias[col] : 0.0f;
#pragma unroll
            for (int r = 0; r < 8; ++r) {
                const int row = m0 + wr * 32 + mt * 16 + laneH * 8 + r;
                float v = acc[mt][nt][r] + bv;
                if (flags & 1) v = gelu_tanh(v);
                if (flags & 2) v += resid[(size_t)row * N + col];
                if (flags & 4) outF[(size_t)row * N + col] = v;
                else           outB[(size_t)row * N + col] = (bf16)v;
            }
        }
    }
}

// ---------------------------------------------------------------------------
// Flash attention (bf16 in, bf16 out, f32 softmax/accum).
// One block = one (b,h) and 128 query rows; 8 waves x 16 rows each.
// kvLen % 32 == 0. O row stride = C_.
// ---------------------------------------------------------------------------
__global__ __launch_bounds__(256)
void flash_attn_wmma(const bf16* __restrict__ Q, int qStride,
                     const bf16* __restrict__ Kp, int kStride,
                     const bf16* __restrict__ Vp, int vStride,
                     bf16* __restrict__ O,
                     int kvLen, int causal, float scale)
{
    __shared__ bf16 Ks[32][64];        // K chunk, [s][d]
    __shared__ bf16 Vt[64][40];        // V chunk transposed, [d][s] (+pad)
    __shared__ bf16 Ps[8][16][32];     // per-wave P staging (C-layout -> A-layout)

    const int tid   = threadIdx.x;
    const int w     = tid >> 5, lane = tid & 31;
    const int laneM = lane & 15, laneH = lane >> 4;
    const int b     = blockIdx.y / H_;
    const int h     = blockIdx.y % H_;
    const int qblk  = blockIdx.x * 128;
    const int qbase = qblk + w * 16;

    // Q fragments for this wave's 16 rows (D=64 -> two K=32 steps)
    bf16x16 qf[2];
#pragma unroll
    for (int kk = 0; kk < 2; ++kk) {
        const bf16* qp = Q + (size_t)(b * T_ + qbase + laneM) * qStride + h * D_ + kk * 32 + laneH * 8;
        qf[kk] = make_frag(qp, qp + 16);
    }

    f32x8 o[4];
    float mrow[8], lrow[8];
#pragma unroll
    for (int nt = 0; nt < 4; ++nt)
#pragma unroll
        for (int r = 0; r < 8; ++r) o[nt][r] = 0.0f;
#pragma unroll
    for (int r = 0; r < 8; ++r) { mrow[r] = -1e30f; lrow[r] = 0.0f; }

    const int li  = tid >> 3;  // 0..31 : chunk row
    const int seg = tid & 7;   // 8-element segment of D

    int sEnd = kvLen;
    if (causal) { int lim = qblk + 128; sEnd = lim < kvLen ? lim : kvLen; }

    for (int s0 = 0; s0 < sEnd; s0 += 32) {
        // K chunk: async DMA straight into LDS; V chunk: VGPR transpose
        {
            const bf16* kp = Kp + (size_t)(b * kvLen + s0 + li) * kStride + h * D_ + seg * 8;
            async_copy16(kp, &Ks[li][seg * 8]);
            const bf16* vp = Vp + (size_t)(b * kvLen + s0 + li) * vStride + h * D_ + seg * 8;
            __align__(16) bf16 tmp[8];
            *reinterpret_cast<int4*>(tmp) = *reinterpret_cast<const int4*>(vp);
#pragma unroll
            for (int j = 0; j < 8; ++j) Vt[seg * 8 + j][li] = tmp[j];
        }
        wait_async0();
        __syncthreads();

        // scores: 16 rows x 32 kv cols via 2 N-tiles, K-dim = D in 2 steps
        f32x8 sc[2];
#pragma unroll
        for (int j = 0; j < 2; ++j) {
#pragma unroll
            for (int r = 0; r < 8; ++r) sc[j][r] = 0.0f;
#pragma unroll
            for (int kk = 0; kk < 2; ++kk) {
                const bf16* kb = &Ks[j * 16 + laneM][kk * 32 + laneH * 8];
                bf16x16 bfrag = make_frag(kb, kb + 16);
                sc[j] = wmma_bf16(qf[kk], bfrag, sc[j]);
            }
        }
        // scale + causal mask
#pragma unroll
        for (int j = 0; j < 2; ++j) {
            const int scol = s0 + j * 16 + laneM;
#pragma unroll
            for (int r = 0; r < 8; ++r) {
                float v = sc[j][r] * scale;
                if (causal && scol > (qbase + laneH * 8 + r)) v = -1e30f;
                sc[j][r] = v;
            }
        }
        // online softmax: row stats live per (r, half-wave); reduce over 16 lanes
        float corr[8];
#pragma unroll
        for (int r = 0; r < 8; ++r) {
            float rmax = fmaxf(sc[0][r], sc[1][r]);
#pragma unroll
            for (int off = 8; off >= 1; off >>= 1) rmax = fmaxf(rmax, __shfl_xor(rmax, off, 16));
            float mnew = fmaxf(mrow[r], rmax);
            corr[r] = expf(mrow[r] - mnew);
            float p0 = expf(sc[0][r] - mnew);
            float p1 = expf(sc[1][r] - mnew);
            sc[0][r] = p0; sc[1][r] = p1;
            float part = p0 + p1;
#pragma unroll
            for (int off = 8; off >= 1; off >>= 1) part += __shfl_xor(part, off, 16);
            lrow[r] = lrow[r] * corr[r] + part;
            mrow[r] = mnew;
        }
#pragma unroll
        for (int nt = 0; nt < 4; ++nt)
#pragma unroll
            for (int r = 0; r < 8; ++r) o[nt][r] *= corr[r];

        // P: C-layout -> A-layout through per-wave LDS (wave-local, in-order)
#pragma unroll
        for (int j = 0; j < 2; ++j)
#pragma unroll
            for (int r = 0; r < 8; ++r)
                Ps[w][laneH * 8 + r][j * 16 + laneM] = (bf16)sc[j][r];
        const bf16* pb = &Ps[w][laneM][0];
        bf16x16 pf = make_frag(pb + laneH * 8, pb + 16 + laneH * 8);

        // O += P @ V  (K-dim = 32 kv rows, N = 64 D in 4 tiles)
#pragma unroll
        for (int nt = 0; nt < 4; ++nt) {
            const bf16* vb = &Vt[nt * 16 + laneM][laneH * 8];
            bf16x16 vf = make_frag(vb, vb + 16);
            o[nt] = wmma_bf16(pf, vf, o[nt]);
        }
        __syncthreads();
    }

    // normalize + merge heads: out[b*T+q][h*64 + d]
#pragma unroll
    for (int nt = 0; nt < 4; ++nt) {
#pragma unroll
        for (int r = 0; r < 8; ++r) {
            const int qr = qbase + laneH * 8 + r;
            float v = o[nt][r] / lrow[r];
            O[(size_t)(b * T_ + qr) * C_ + h * D_ + nt * 16 + laneM] = (bf16)v;
        }
    }
}

// ---------------------------------------------------------------------------
// LayerNorm (f32 in, bf16 out). One block per row.
// ---------------------------------------------------------------------------
__global__ __launch_bounds__(256)
void layernorm_to_bf16(const float* __restrict__ X, const float* __restrict__ g,
                       const float* __restrict__ bb, bf16* __restrict__ Y, int Cc)
{
    __shared__ float red[256];
    const int tid = threadIdx.x;
    const size_t base = (size_t)blockIdx.x * Cc;

    float s = 0.0f;
    for (int c = tid; c < Cc; c += 256) s += X[base + c];
    red[tid] = s; __syncthreads();
    for (int st = 128; st > 0; st >>= 1) { if (tid < st) red[tid] += red[tid + st]; __syncthreads(); }
    const float mu = red[0] / Cc;
    __syncthreads();

    float v = 0.0f;
    for (int c = tid; c < Cc; c += 256) { float d = X[base + c] - mu; v += d * d; }
    red[tid] = v; __syncthreads();
    for (int st = 128; st > 0; st >>= 1) { if (tid < st) red[tid] += red[tid + st]; __syncthreads(); }
    const float rinv = rsqrtf(red[0] / Cc + 1e-5f);

    for (int c = tid; c < Cc; c += 256)
        Y[base + c] = (bf16)((X[base + c] - mu) * rinv * g[c] + bb[c]);
}

// Wt[n][k] = (bf16) W[k][n]; tiled via LDS. Kd%32==0, Nd%32==0.
__global__ __launch_bounds__(256)
void transpose_cast(const float* __restrict__ W, bf16* __restrict__ Wt, int Kd, int Nd)
{
    __shared__ float tile[32][33];
    const int n0 = blockIdx.x * 32, k0 = blockIdx.y * 32;
    const int tx = threadIdx.x & 31, ty = threadIdx.x >> 5;  // 32x8
#pragma unroll
    for (int i = 0; i < 32; i += 8)
        tile[ty + i][tx] = W[(size_t)(k0 + ty + i) * Nd + n0 + tx];
    __syncthreads();
#pragma unroll
    for (int i = 0; i < 32; i += 8)
        Wt[(size_t)(n0 + ty + i) * Kd + k0 + tx] = (bf16)tile[tx][ty + i];
}

__global__ __launch_bounds__(256)
void cast_f32_bf16(const float* __restrict__ X, bf16* __restrict__ Y, size_t n)
{
    size_t idx = (size_t)blockIdx.x * blockDim.x + threadIdx.x;
    if (idx < n) Y[idx] = (bf16)X[idx];
}

// ---------------------------------------------------------------------------
extern "C" void kernel_launch(void* const* d_in, const int* in_sizes, int n_in,
                              void* d_out, int out_size, void* d_ws, size_t ws_size,
                              hipStream_t stream)
{
    (void)in_sizes; (void)n_in; (void)out_size; (void)ws_size;
    const float* x       = (const float*)d_in[0];
    const float* ximg    = (const float*)d_in[1];
    const float* ln1_g   = (const float*)d_in[2];
    const float* ln1_b   = (const float*)d_in[3];
    const float* ln2_g   = (const float*)d_in[4];
    const float* ln2_b   = (const float*)d_in[5];
    const float* W_attn  = (const float*)d_in[6];
    const float* b_attn  = (const float*)d_in[7];
    const float* W_aproj = (const float*)d_in[8];
    const float* b_aproj = (const float*)d_in[9];
    const float* Wq      = (const float*)d_in[10];
    const float* bq      = (const float*)d_in[11];
    const float* Wk      = (const float*)d_in[12];
    const float* bk      = (const float*)d_in[13];
    const float* Wv      = (const float*)d_in[14];
    const float* bv      = (const float*)d_in[15];
    const float* Wcproj  = (const float*)d_in[16];
    const float* bcproj  = (const float*)d_in[17];
    const float* W_fc    = (const float*)d_in[18];
    const float* b_fc    = (const float*)d_in[19];
    const float* W_mproj = (const float*)d_in[20];
    const float* b_mproj = (const float*)d_in[21];
    float* out = (float*)d_out;

    const int BT  = B_ * T_;    // 8192
    const int BTI = B_ * TI_;   // 2048

    char* ws = (char*)d_ws;
    size_t off = 0;
    auto alloc = [&](size_t bytes) -> void* {
        void* p = ws + off; off += (bytes + 255) & ~(size_t)255; return p;
    };

    bf16* wt_attn  = (bf16*)alloc((size_t)3 * C_ * C_ * sizeof(bf16));
    bf16* wt_aproj = (bf16*)alloc((size_t)C_ * C_ * sizeof(bf16));
    bf16* wt_q     = (bf16*)alloc((size_t)C_ * C_ * sizeof(bf16));
    bf16* wt_k     = (bf16*)alloc((size_t)C_ * C_ * sizeof(bf16));
    bf16* wt_v     = (bf16*)alloc((size_t)C_ * C_ * sizeof(bf16));
    bf16* wt_cproj = (bf16*)alloc((size_t)C_ * C_ * sizeof(bf16));
    bf16* wt_fc    = (bf16*)alloc((size_t)FF_ * C_ * sizeof(bf16));
    bf16* wt_mproj = (bf16*)alloc((size_t)C_ * FF_ * sizeof(bf16));
    bf16* act16    = (bf16*)alloc((size_t)BT * C_ * sizeof(bf16));     // ln outputs
    bf16* bigbuf   = (bf16*)alloc((size_t)BT * FF_ * sizeof(bf16));    // qkv, then fc out
    bf16* pbuf     = (bf16*)alloc((size_t)BT * C_ * sizeof(bf16));     // attn outs
    bf16* qbuf     = (bf16*)alloc((size_t)BT * C_ * sizeof(bf16));     // cross Q
    bf16* img16    = (bf16*)alloc((size_t)BTI * C_ * sizeof(bf16));
    bf16* k2       = (bf16*)alloc((size_t)BTI * C_ * sizeof(bf16));
    bf16* v2       = (bf16*)alloc((size_t)BTI * C_ * sizeof(bf16));
    float* x1      = (float*)alloc((size_t)BT * C_ * sizeof(float));

    const dim3 blk(256);
    const float scale = 0.125f;  // 1/sqrt(64)

    // weight prep (Wt[n][k] = W[k][n], bf16)
    transpose_cast<<<dim3(3 * C_ / 32, C_ / 32), blk, 0, stream>>>(W_attn,  wt_attn,  C_,  3 * C_);
    transpose_cast<<<dim3(C_ / 32,     C_ / 32), blk, 0, stream>>>(W_aproj, wt_aproj, C_,  C_);
    transpose_cast<<<dim3(C_ / 32,     C_ / 32), blk, 0, stream>>>(Wq,      wt_q,     C_,  C_);
    transpose_cast<<<dim3(C_ / 32,     C_ / 32), blk, 0, stream>>>(Wk,      wt_k,     C_,  C_);
    transpose_cast<<<dim3(C_ / 32,     C_ / 32), blk, 0, stream>>>(Wv,      wt_v,     C_,  C_);
    transpose_cast<<<dim3(C_ / 32,     C_ / 32), blk, 0, stream>>>(Wcproj,  wt_cproj, C_,  C_);
    transpose_cast<<<dim3(FF_ / 32,    C_ / 32), blk, 0, stream>>>(W_fc,    wt_fc,    C_,  FF_);
    transpose_cast<<<dim3(C_ / 32,    FF_ / 32), blk, 0, stream>>>(W_mproj, wt_mproj, FF_, C_);
    cast_f32_bf16<<<(BTI * C_) / 256, blk, 0, stream>>>(ximg, img16, (size_t)BTI * C_);

    // --- causal self-attention ---
    layernorm_to_bf16<<<BT, blk, 0, stream>>>(x, ln1_g, ln1_b, act16, C_);
    gemm_bf16_wmma<<<dim3(3 * C_ / 128, BT / 128), blk, 0, stream>>>(
        act16, wt_attn, b_attn, nullptr, nullptr, bigbuf, BT, 3 * C_, C_, 0);
    flash_attn_wmma<<<dim3(T_ / 128, B_ * H_), blk, 0, stream>>>(
        bigbuf, 3 * C_, bigbuf + C_, 3 * C_, bigbuf + 2 * C_, 3 * C_, pbuf, T_, 1, scale);
    gemm_bf16_wmma<<<dim3(C_ / 128, BT / 128), blk, 0, stream>>>(
        pbuf, wt_aproj, b_aproj, x, x1, nullptr, BT, C_, C_, 2 | 4);       // x1 = x + sa

    // --- cross-attention ---
    layernorm_to_bf16<<<BT, blk, 0, stream>>>(x1, ln1_g, ln1_b, act16, C_);
    gemm_bf16_wmma<<<dim3(C_ / 128, BT / 128), blk, 0, stream>>>(
        act16, wt_q, bq, nullptr, nullptr, qbuf, BT, C_, C_, 0);
    gemm_bf16_wmma<<<dim3(C_ / 128, BTI / 128), blk, 0, stream>>>(
        img16, wt_k, bk, nullptr, nullptr, k2, BTI, C_, C_, 0);
    gemm_bf16_wmma<<<dim3(C_ / 128, BTI / 128), blk, 0, stream>>>(
        img16, wt_v, bv, nullptr, nullptr, v2, BTI, C_, C_, 0);
    flash_attn_wmma<<<dim3(T_ / 128, B_ * H_), blk, 0, stream>>>(
        qbuf, C_, k2, C_, v2, C_, pbuf, TI_, 0, scale);
    gemm_bf16_wmma<<<dim3(C_ / 128, BT / 128), blk, 0, stream>>>(
        pbuf, wt_cproj, bcproj, x1, out, nullptr, BT, C_, C_, 2 | 4);      // out = x1 + ca

    // --- MLP ---
    layernorm_to_bf16<<<BT, blk, 0, stream>>>(out, ln2_g, ln2_b, act16, C_);
    gemm_bf16_wmma<<<dim3(FF_ / 128, BT / 128), blk, 0, stream>>>(
        act16, wt_fc, b_fc, nullptr, nullptr, bigbuf, BT, FF_, C_, 1);     // gelu
    gemm_bf16_wmma<<<dim3(C_ / 128, BT / 128), blk, 0, stream>>>(
        bigbuf, wt_mproj, b_mproj, out, out, nullptr, BT, C_, FF_, 2 | 4); // out += mlp
}
